// FieldConditionedEquivariantAtomAttention_80805514707446
// MI455X (gfx1250) — matrix operand order repaired and checked by reading, not compiled, not run on hardware
//
#include <hip/hip_runtime.h>
#include <hip/hip_bf16.h>
#include <math.h>

// ---------------------------------------------------------------------------
// MI455X / gfx1250 FieldConditionedEquivariantAtomAttention
// Dense layers: v_wmma_f32_16x16x32_f16, fragment-order LDS staging so every
// WMMA operand is loaded with 2x ds_load_b128 per matrix (no scalar gathers).
// ---------------------------------------------------------------------------

typedef __attribute__((ext_vector_type(16))) _Float16 v16h;
typedef __attribute__((ext_vector_type(8)))  _Float16 v8h;
typedef __attribute__((ext_vector_type(4)))  _Float16 v4h;
typedef __attribute__((ext_vector_type(8)))  float    v8f;

#define B_      8
#define N_      64
#define NE_     128
#define D_      160
#define FA_     64
#define FB_     64
#define ZEMB_   32
#define TPWSZ_  9216
#define CUTF_   6.0f
#define SQ3_    1.7320508075688772f
#define ALPHA_  0.10206207261596575f   /* 1/sqrt(96) */
#define PI_     3.14159265358979323846f

__device__ __forceinline__ float siluf(float x) { return x / (1.f + __expf(-x)); }
__device__ __forceinline__ float sigm(float x)  { return 1.f / (1.f + __expf(-x)); }

// K permutation within a 32-wide chunk so the 16x32 f16 A-fragment
// (ISA 7.12.2: lanes 0-15 -> K{0..7,16..23}, lanes 16-31 -> K{8..15,24..31})
// becomes two contiguous b128 LDS loads at [half*16].
__device__ __forceinline__ int kperm(int k) {
    return (k & 7) | ((k & 8) << 1) | ((k & 16) >> 1);
}
__device__ __forceinline__ v16h cat8(v8h lo, v8h hi) {
    return __builtin_shufflevector(lo, hi, 0,1,2,3,4,5,6,7,8,9,10,11,12,13,14,15);
}

// ---------------------------------------------------------------------------
// Tiled GEMM: C[M,N] = act(A[M,K] @ W + bias), weights given as WT[N,K] f16.
// Tile 64x128, 8 waves (4 row-blocks x 2 col-blocks), BK=32.
// Requires 64|M, 128|N, 32|K (true for all shapes here).
// ---------------------------------------------------------------------------
template<int ACT>
__global__ __launch_bounds__(256)
void gemm_f16_wmma(const float* __restrict__ A, const _Float16* __restrict__ BT,
                   const float* __restrict__ bias, float* __restrict__ C,
                   int M, int Ncol, int K)
{
    __shared__ _Float16 Asw[64][40];    // K-permuted, 80B rows (16B aligned)
    __shared__ _Float16 Bsw[128][40];   // fragment-order: [n][k]

    const int tid   = threadIdx.x;
    const int lane  = tid & 31;
    const int wid   = tid >> 5;
    const int wm    = wid & 3;
    const int wn    = wid >> 2;
    const int mlane = lane & 15;
    const int half  = lane >> 4;
    const int m0    = blockIdx.y * 64;
    const int n0    = blockIdx.x * 128;

    v8f acc[4];
    #pragma unroll
    for (int t = 0; t < 4; ++t)
        #pragma unroll
        for (int j = 0; j < 8; ++j) acc[t][j] = 0.f;

    for (int kk = 0; kk < K; kk += 32) {
        // ---- stage A: 64x32 f32 -> f16 K-permuted (float4 load, b64 store)
        #pragma unroll
        for (int i = 0; i < 2; ++i) {
            int task = i * 256 + tid;            // 512 tasks of 4 elements
            int r  = task >> 3;
            int k0 = (task & 7) * 4;
            float4 f = *(const float4*)&A[(size_t)(m0 + r) * K + kk + k0];
            v4h h; h[0] = (_Float16)f.x; h[1] = (_Float16)f.y;
                   h[2] = (_Float16)f.z; h[3] = (_Float16)f.w;
            *(v4h*)&Asw[r][kperm(k0)] = h;
        }
        // ---- stage B: straight b128 copies of WT rows
        {
            int r = tid >> 1, h = tid & 1;
            const _Float16* src = &BT[(size_t)(n0 + r) * K + kk + h * 16];
            *(v8h*)&Bsw[r][h * 16]     = *(const v8h*)src;
            *(v8h*)&Bsw[r][h * 16 + 8] = *(const v8h*)(src + 8);
        }
        if (kk + 32 < K)
            __builtin_prefetch(&BT[(size_t)(n0 + (tid >> 1)) * K + kk + 32], 0, 1);
        __syncthreads();

        v16h a = cat8(*(const v8h*)&Asw[wm * 16 + mlane][half * 16],
                      *(const v8h*)&Asw[wm * 16 + mlane][half * 16 + 8]);
        #pragma unroll
        for (int t = 0; t < 4; ++t) {
            int cb = wn * 64 + t * 16 + mlane;
            v16h b = cat8(*(const v8h*)&Bsw[cb][half * 8],
                          *(const v8h*)&Bsw[cb][16 + half * 8]);
            acc[t] = __builtin_amdgcn_wmma_f32_16x16x32_f16(
                false, a, false, b, (short)0, acc[t], false, false);
        }
        __syncthreads();
    }

    #pragma unroll
    for (int t = 0; t < 4; ++t) {
        int col = n0 + wn * 64 + t * 16 + mlane;
        float bv = bias ? bias[col] : 0.f;
        #pragma unroll
        for (int j = 0; j < 8; ++j) {
            int row = m0 + wm * 16 + half * 8 + j;
            float v = acc[t][j] + bv;
            if (ACT == 1) v = siluf(v);
            C[(size_t)row * Ncol + col] = v;
        }
    }
}

// ---------------------------------------------------------------------------
// Tiled transpose + f32->f16 convert: WT[n][k] = (f16)W[k][n]
// ---------------------------------------------------------------------------
__global__ __launch_bounds__(256)
void transpose_f32_to_f16(const float* __restrict__ W, _Float16* __restrict__ WT,
                          int K, int Ncol)
{
    __shared__ float tile[32][33];
    const int kb = blockIdx.y * 32, nb = blockIdx.x * 32;
    const int tx = threadIdx.x & 31, ty = threadIdx.x >> 5;   // 32x8
    #pragma unroll
    for (int i = 0; i < 32; i += 8)
        tile[ty + i][tx] = W[(size_t)(kb + ty + i) * Ncol + nb + tx];
    __syncthreads();
    #pragma unroll
    for (int i = 0; i < 32; i += 8)
        WT[(size_t)(nb + ty + i) * K + kb + tx] = (_Float16)tile[tx][ty + i];
}

// score-W0 rows [96:256) + [320:384) gathered, transposed, converted:
// w0T[n][k] = (f16) sW0[src(k)][n],  k in [0,224)
__global__ __launch_bounds__(256)
void repack_w0T(const float* __restrict__ sW0, _Float16* __restrict__ w0T)
{
    int idx = blockIdx.x * 256 + threadIdx.x;   // 256*224 total
    int n = idx / 224, k = idx % 224;
    int src = (k < 160) ? (k + 96) : (k + 160);
    w0T[idx] = (_Float16)sW0[src * 256 + n];
}

// ---------------------------------------------------------------------------
// prep: geometry, RBF, z-emb, invariant feats -> value_in / node_in
// ---------------------------------------------------------------------------
__global__ __launch_bounds__(128)
void prep_kernel(const float* __restrict__ h_full, const int* __restrict__ z,
                 const float* __restrict__ pos, const unsigned char* __restrict__ mask,
                 const float* __restrict__ fal, const float* __restrict__ fab,
                 const float* __restrict__ z_emb, const int* __restrict__ pabs,
                 float* __restrict__ vin, float* __restrict__ nin,
                 float* __restrict__ sh1b, float* __restrict__ cutb,
                 float* __restrict__ validb)
{
    const int row = blockIdx.x;           // b*64 + n
    const int b = row >> 6, n = row & 63;
    const int t = threadIdx.x;
    const int ai = pabs[0];

    const float ax = pos[(b * 64 + ai) * 3 + 0];
    const float ay = pos[(b * 64 + ai) * 3 + 1];
    const float az = pos[(b * 64 + ai) * 3 + 2];
    const float rx = pos[row * 3 + 0] - ax;
    const float ry = pos[row * 3 + 1] - ay;
    const float rz = pos[row * 3 + 2] - az;
    const float r  = sqrtf(fmaxf(rx * rx + ry * ry + rz * rz, 1e-16f));
    const float iv = 1.f / fmaxf(r, 1e-8f);
    const float ux = rx * iv, uy = ry * iv, uz = rz * iv;
    const float rmin = fminf(r, CUTF_);

    float* vrow = &vin[(size_t)row * 192];
    float* nrow = &nin[(size_t)row * 224];

    if (t < 32) {
        float zv = z_emb[(size_t)z[row] * ZEMB_ + t];
        vrow[t] = zv;        nrow[96 + t] = zv;
        const float delta = CUTF_ / 31.f;
        const float gamma = 1.f / (delta * delta + 1e-12f);
        float cen = CUTF_ * (float)t / 31.f;
        float dd  = rmin - cen;
        float rrv = __expf(-gamma * dd * dd);
        vrow[32 + t] = rrv;  nrow[128 + t] = rrv;
        float x0 = h_full[(size_t)row * D_ + 64 + t * 3 + 0];
        float x1 = h_full[(size_t)row * D_ + 64 + t * 3 + 1];
        float x2 = h_full[(size_t)row * D_ + 64 + t * 3 + 2];
        nrow[64 + t] = sqrtf((x0 * x0 + x1 * x1 + x2 * x2) / 3.f + 1e-8f);
    }
    if (t < 64) {
        float fv = fal[(size_t)row * FA_ + t];
        vrow[64 + t]  = fv;
        nrow[160 + t] = fv;
        vrow[128 + t] = fab[b * FB_ + t];
        nrow[t] = h_full[(size_t)row * D_ + t];
    }
    if (t == 0) {
        sh1b[row * 3 + 0] = SQ3_ * uy;
        sh1b[row * 3 + 1] = SQ3_ * uz;
        sh1b[row * 3 + 2] = SQ3_ * ux;
        float inside = (r <= CUTF_) ? 1.f : 0.f;
        cutb[row]  = 0.5f * (__cosf(PI_ * r / CUTF_) + 1.f) * inside;
        validb[row] = ((mask[row] != 0) && (n != ai) && (r <= CUTF_)) ? 1.f : 0.f;
    }
}

__global__ __launch_bounds__(256)
void batchproj_kernel(const float* __restrict__ h_full, const float* __restrict__ fab,
                      const float* __restrict__ sW0, const int* __restrict__ pabs,
                      float* __restrict__ bp)
{
    __shared__ float ia[96];
    const int b = blockIdx.x, t = threadIdx.x;
    const int ai = pabs[0];
    const float* h = &h_full[(size_t)(b * 64 + ai) * D_];
    if (t < 64) ia[t] = h[t];
    if (t < 32) {
        float x = h[64 + t * 3], y = h[64 + t * 3 + 1], zc = h[64 + t * 3 + 2];
        ia[64 + t] = sqrtf((x * x + y * y + zc * zc) / 3.f + 1e-8f);
    }
    __syncthreads();
    float acc = 0.f;
    for (int k = 0; k < 96; ++k) acc += ia[k] * sW0[k * 256 + t];
    for (int k = 0; k < 64; ++k) acc += fab[b * FB_ + k] * sW0[(384 + k) * 256 + t];
    bp[b * 256 + t] = acc;
}

// ---------------------------------------------------------------------------
// Fused score kernel: block = (b,e), rows n=0..63.
// silu(h1) built K-permuted in f16 LDS; h2 GEMM via WMMA against f16 W1T;
// b1+silu+W2-dot folded from accumulators via ds_add_f32; emits the gate.
// ---------------------------------------------------------------------------
__global__ __launch_bounds__(256)
void score_kernel(const float* __restrict__ bp,  const float* __restrict__ np,
                  const float* __restrict__ ep,  const float* __restrict__ sb0,
                  const _Float16* __restrict__ sW1T, const float* __restrict__ sb1,
                  const float* __restrict__ sW2, const float* __restrict__ sb2,
                  const float* __restrict__ cutb, const float* __restrict__ validb,
                  float* __restrict__ gate)
{
    __shared__ _Float16 h1s[64][264];   // 528B rows (16B aligned), K-permuted/32
    __shared__ _Float16 Bsw[256][40];   // W1T slice in fragment order
    __shared__ float    sscore[64];

    const int b = blockIdx.y, e = blockIdx.x;
    const int tid  = threadIdx.x;
    const int lane = tid & 31, wid = tid >> 5;
    const int mlane = lane & 15, half = lane >> 4;
    const int wm = wid & 3, wn = wid >> 2;

    if (tid < 64) sscore[tid] = 0.f;

    #pragma unroll 4
    for (int i = 0; i < 16; ++i) {              // 4096 tasks of 4 columns
        int task = i * 256 + tid;
        int r  = task >> 6;
        int c0 = (task & 63) * 4;
        float4 npv = *(const float4*)&np[(size_t)(b * 64 + r) * 256 + c0];
        float4 bpv = *(const float4*)&bp[b * 256 + c0];
        float4 epv = *(const float4*)&ep[(size_t)e * 256 + c0];
        float4 b0v = *(const float4*)&sb0[c0];
        v4h h;
        h[0] = (_Float16)siluf(npv.x + bpv.x + epv.x + b0v.x);
        h[1] = (_Float16)siluf(npv.y + bpv.y + epv.y + b0v.y);
        h[2] = (_Float16)siluf(npv.z + bpv.z + epv.z + b0v.z);
        h[3] = (_Float16)siluf(npv.w + bpv.w + epv.w + b0v.w);
        *(v4h*)&h1s[r][(c0 & ~31) | kperm(c0 & 31)] = h;
    }
    __syncthreads();

    v8f acc[8];
    #pragma unroll
    for (int t = 0; t < 8; ++t)
        #pragma unroll
        for (int j = 0; j < 8; ++j) acc[t][j] = 0.f;

    for (int kk = 0; kk < 256; kk += 32) {
        {   // stage W1T slice: 256 rows x 32 f16, one row per thread (b128 x4)
            const _Float16* src = &sW1T[(size_t)tid * 256 + kk];
            *(v8h*)&Bsw[tid][0]  = *(const v8h*)src;
            *(v8h*)&Bsw[tid][8]  = *(const v8h*)(src + 8);
            *(v8h*)&Bsw[tid][16] = *(const v8h*)(src + 16);
            *(v8h*)&Bsw[tid][24] = *(const v8h*)(src + 24);
        }
        __syncthreads();

        v16h a = cat8(*(const v8h*)&h1s[wm * 16 + mlane][kk + half * 16],
                      *(const v8h*)&h1s[wm * 16 + mlane][kk + half * 16 + 8]);
        #pragma unroll
        for (int t = 0; t < 8; ++t) {
            int cb = wn * 128 + t * 16 + mlane;
            v16h bf = cat8(*(const v8h*)&Bsw[cb][half * 8],
                           *(const v8h*)&Bsw[cb][16 + half * 8]);
            acc[t] = __builtin_amdgcn_wmma_f32_16x16x32_f16(
                false, a, false, bf, (short)0, acc[t], false, false);
        }
        __syncthreads();
    }

    float racc[8];
    #pragma unroll
    for (int j = 0; j < 8; ++j) racc[j] = 0.f;
    #pragma unroll
    for (int t = 0; t < 8; ++t) {
        int col = wn * 128 + t * 16 + mlane;
        float b1v = sb1[col], w2v = sW2[col];
        #pragma unroll
        for (int j = 0; j < 8; ++j)
            racc[j] += siluf(acc[t][j] + b1v) * w2v;
    }
    #pragma unroll
    for (int j = 0; j < 8; ++j)
        atomicAdd(&sscore[wm * 16 + half * 8 + j], racc[j]);   // ds_add_f32
    __syncthreads();

    if (tid < 64) {
        int grow = b * 64 + tid;
        float g = sigm(sscore[tid] + sb2[0]) * cutb[grow] * validb[grow];
        gate[((size_t)(b * NE_) + e) * 64 + tid] = g;
    }
}

// ---------------------------------------------------------------------------
// Tensor-product contraction: per (b,n) row of tp_w -> values (160), fp32 VALU
// ---------------------------------------------------------------------------
__global__ __launch_bounds__(192)
void tp_kernel(const float* __restrict__ h_full, const float* __restrict__ tpw,
               const float* __restrict__ sh1b, float* __restrict__ vals)
{
    __shared__ float s1[64], v1[32][3], d12[32], sh1[3];
    const int row = blockIdx.x, t = threadIdx.x;
    if (t < 64) s1[t] = h_full[(size_t)row * D_ + t];
    if (t < 96) v1[t / 3][t % 3] = h_full[(size_t)row * D_ + 64 + t];
    if (t < 3)  sh1[t] = sh1b[row * 3 + t];
    __syncthreads();
    if (t < 32)
        d12[t] = (v1[t][0] * sh1[0] + v1[t][1] * sh1[1] + v1[t][2] * sh1[2]) * (1.f / SQ3_);
    __syncthreads();

    const float* w = &tpw[(size_t)row * TPWSZ_];
    if (t < 64) {
        float o = 0.f;
        for (int u = 0; u < 64; ++u) o += w[u * 64 + t] * s1[u];          // w_00_0
        for (int u = 0; u < 32; ++u) o += w[7168 + u * 64 + t] * d12[u];  // w_11_0
        vals[(size_t)row * D_ + t] = o * ALPHA_;
    } else if (t < 160) {
        int idx = t - 64, ww = idx / 3, x = idx % 3;
        float o01 = 0.f, o10 = 0.f;
        for (int u = 0; u < 64; ++u) o01 += w[4096 + u * 32 + ww] * s1[u];    // w_01_1
        for (int u = 0; u < 32; ++u) o10 += w[6144 + u * 32 + ww] * v1[u][x]; // w_10_1
        vals[(size_t)row * D_ + t] = (o01 * sh1[x] + o10) * ALPHA_;
    }
}

// ---------------------------------------------------------------------------
// Gated aggregation + invariants of agg + fab concat -> out_in (1024x160)
// ---------------------------------------------------------------------------
__global__ __launch_bounds__(192)
void agg_kernel(const float* __restrict__ gate, const float* __restrict__ vals,
                const float* __restrict__ fab, float* __restrict__ oin)
{
    __shared__ float g[64], ag[160];
    const int b = blockIdx.y, e = blockIdx.x, t = threadIdx.x;
    if (t < 64) g[t] = gate[((size_t)(b * NE_) + e) * 64 + t];
    __syncthreads();
    float norm = 0.f;
    for (int n = 0; n < 64; ++n) norm += g[n];
    norm = fmaxf(norm, 1e-8f);
    if (t < 160) {
        float a = 0.f;
        for (int n = 0; n < 64; ++n) a += g[n] * vals[(size_t)(b * 64 + n) * D_ + t];
        ag[t] = a / norm;
    }
    __syncthreads();
    const size_t row = (size_t)b * NE_ + e;
    if (t < 64) oin[row * 160 + t] = ag[t];
    if (t < 32) {
        float x = ag[64 + t * 3], y = ag[64 + t * 3 + 1], zc = ag[64 + t * 3 + 2];
        oin[row * 160 + 64 + t] = sqrtf((x * x + y * y + zc * zc) / 3.f + 1e-8f);
    }
    if (t >= 96 && t < 160) oin[row * 160 + t] = fab[b * FB_ + (t - 96)];
}

// ---------------------------------------------------------------------------
// launch
// ---------------------------------------------------------------------------
extern "C" void kernel_launch(void* const* d_in, const int* in_sizes, int n_in,
                              void* d_out, int out_size, void* d_ws, size_t ws_size,
                              hipStream_t stream)
{
    (void)in_sizes; (void)n_in; (void)out_size; (void)ws_size;

    const float*         h_full = (const float*)d_in[0];
    const int*           z      = (const int*)d_in[1];
    const float*         pos    = (const float*)d_in[2];
    const unsigned char* mask   = (const unsigned char*)d_in[3];
    const float*         e_feat = (const float*)d_in[4];
    const float*         fal    = (const float*)d_in[5];
    const float*         fab    = (const float*)d_in[6];
    const float*         z_emb  = (const float*)d_in[7];
    const float* vW0 = (const float*)d_in[8];  const float* vb0 = (const float*)d_in[9];
    const float* vW1 = (const float*)d_in[10]; const float* vb1 = (const float*)d_in[11];
    const float* vW2 = (const float*)d_in[12]; const float* vb2 = (const float*)d_in[13];
    const float* sW0 = (const float*)d_in[14]; const float* sb0 = (const float*)d_in[15];
    const float* sW1 = (const float*)d_in[16]; const float* sb1 = (const float*)d_in[17];
    const float* sW2 = (const float*)d_in[18]; const float* sb2 = (const float*)d_in[19];
    const float* oW0 = (const float*)d_in[20]; const float* ob0 = (const float*)d_in[21];
    const float* oW1 = (const float*)d_in[22]; const float* ob1 = (const float*)d_in[23];
    const float* oW2 = (const float*)d_in[24]; const float* ob2 = (const float*)d_in[25];
    const int*   pabs = (const int*)d_in[26];

    float* ws = (float*)d_ws;
    size_t o = 0;
    float* vin    = ws + o; o += 512 * 192;
    float* nin    = ws + o; o += 512 * 224;
    float* h1v    = ws + o; o += 512 * 256;
    float* h2v    = ws + o; o += 512 * 256;
    float* tpwb   = ws + o; o += (size_t)512 * TPWSZ_;
    float* vals   = ws + o; o += 512 * 160;
    float* nodep  = ws + o; o += 512 * 256;
    float* bprj   = ws + o; o += 8 * 256;
    float* eprj   = ws + o; o += 128 * 256;
    float* gateb  = ws + o; o += 8 * 128 * 64;
    float* sh1b   = ws + o; o += 512 * 3;
    float* cutb   = ws + o; o += 512;
    float* validb = ws + o; o += 512;
    float* oin    = ws + o; o += 1024 * 160;
    float* oh1    = ws + o; o += 1024 * 256;
    float* oh2    = ws + o; o += 1024 * 256;
    o = (o + 31) & ~(size_t)31;                         // 128B-align f16 pool

    _Float16* hp = (_Float16*)(ws + o);
    size_t ho = 0;
    _Float16* vW0T   = hp + ho; ho += (size_t)256 * 192;
    _Float16* vW1T   = hp + ho; ho += (size_t)256 * 256;
    _Float16* vW2T   = hp + ho; ho += (size_t)9216 * 256;
    _Float16* w0catT = hp + ho; ho += (size_t)256 * 224;
    _Float16* WeT    = hp + ho; ho += (size_t)256 * 64;
    _Float16* sW1T   = hp + ho; ho += (size_t)256 * 256;
    _Float16* oW0T   = hp + ho; ho += (size_t)256 * 160;
    _Float16* oW1T   = hp + ho; ho += (size_t)256 * 256;
    _Float16* oW2T   = hp + ho; ho += (size_t)128 * 256;

    // Stage 0: one-time weight transpose + f16 convert (~10 MB total)
    transpose_f32_to_f16<<<dim3(8, 6),   256, 0, stream>>>(vW0, vW0T, 192, 256);
    transpose_f32_to_f16<<<dim3(8, 8),   256, 0, stream>>>(vW1, vW1T, 256, 256);
    transpose_f32_to_f16<<<dim3(288, 8), 256, 0, stream>>>(vW2, vW2T, 256, 9216);
    transpose_f32_to_f16<<<dim3(8, 2),   256, 0, stream>>>(sW0 + 256 * 256, WeT, 64, 256);
    transpose_f32_to_f16<<<dim3(8, 8),   256, 0, stream>>>(sW1, sW1T, 256, 256);
    transpose_f32_to_f16<<<dim3(8, 5),   256, 0, stream>>>(oW0, oW0T, 160, 256);
    transpose_f32_to_f16<<<dim3(8, 8),   256, 0, stream>>>(oW1, oW1T, 256, 256);
    transpose_f32_to_f16<<<dim3(4, 8),   256, 0, stream>>>(oW2, oW2T, 256, 128);
    repack_w0T<<<224, 256, 0, stream>>>(sW0, w0catT);

    // Stage 1: features / projections
    prep_kernel<<<512, 128, 0, stream>>>(h_full, z, pos, mask, fal, fab, z_emb, pabs,
                                         vin, nin, sh1b, cutb, validb);
    batchproj_kernel<<<8, 256, 0, stream>>>(h_full, fab, sW0, pabs, bprj);

    // Stage 2: value MLP (WMMA)
    gemm_f16_wmma<1><<<dim3(2, 8),  256, 0, stream>>>(vin, vW0T, vb0, h1v, 512, 256, 192);
    gemm_f16_wmma<1><<<dim3(2, 8),  256, 0, stream>>>(h1v, vW1T, vb1, h2v, 512, 256, 256);
    gemm_f16_wmma<0><<<dim3(72, 8), 256, 0, stream>>>(h2v, vW2T, vb2, tpwb, 512, 9216, 256);

    // Stage 3: score projections (WMMA)
    gemm_f16_wmma<0><<<dim3(2, 8), 256, 0, stream>>>(nin, w0catT, nullptr, nodep, 512, 256, 224);
    gemm_f16_wmma<0><<<dim3(2, 2), 256, 0, stream>>>(e_feat, WeT, nullptr, eprj, 128, 256, 64);

    // Stage 4: tensor-product values + fused score/gate (WMMA)
    tp_kernel<<<512, 192, 0, stream>>>(h_full, tpwb, sh1b, vals);
    score_kernel<<<dim3(NE_, B_), 256, 0, stream>>>(bprj, nodep, eprj, sb0, sW1T, sb1,
                                                    sW2, sb2, cutb, validb, gateb);

    // Stage 5: aggregation + out MLP (WMMA)
    agg_kernel<<<dim3(NE_, B_), 192, 0, stream>>>(gateb, vals, fab, oin);
    gemm_f16_wmma<1><<<dim3(2, 16), 256, 0, stream>>>(oin, oW0T, ob0, oh1, 1024, 256, 160);
    gemm_f16_wmma<1><<<dim3(2, 16), 256, 0, stream>>>(oh1, oW1T, ob1, oh2, 1024, 256, 256);
    gemm_f16_wmma<0><<<dim3(1, 16), 256, 0, stream>>>(oh2, oW2T, ob2, (float*)d_out,
                                                      1024, 128, 256);
}
